// BiLSTM_55499567398998
// MI455X (gfx1250) — matrix operand order repaired and checked by reading, not compiled, not run on hardware
//
#include <hip/hip_runtime.h>
#include <hip/hip_bf16.h>
#include <stdint.h>

// ---------------- CDNA5 WMMA types ----------------
typedef __attribute__((ext_vector_type(16))) __bf16 v16bf;
typedef __attribute__((ext_vector_type(8)))  __bf16 v8bf;
typedef __attribute__((ext_vector_type(8)))  float  v8f;

union AF { v16bf v; v8bf h[2]; __bf16 e[16]; };

__device__ __forceinline__ float rcp_f(float x) { return __builtin_amdgcn_rcpf(x); }
__device__ __forceinline__ float sigf(float x)  { return rcp_f(1.0f + __expf(-x)); }
__device__ __forceinline__ float tanh_f(float x) {
#if __has_builtin(__builtin_amdgcn_tanhf)
    return __builtin_amdgcn_tanhf(x);          // CDNA5 v_tanh_f32 (TRANS op)
#else
    float e = __expf(2.0f * x);
    return (e - 1.0f) * rcp_f(e + 1.0f);
#endif
}
__device__ __forceinline__ float leaky(float x) { return x > 0.0f ? x : 0.3f * x; }

// =====================================================================
// Layer-1 BiLSTM: one block = (16-batch tile, direction). H1=50.
// Kpad=64 (2 wmma k-steps), N=200 gates (13 N tiles), 8 waves -> at most
// 2 tiles x 2 k-steps = 4 serial WMMAs per timestep per wave.
// Whh bf16 + x tile + Wih/bias resident in static LDS. Input-gate term
// (x*Wih+b) is computed directly into the WMMA C operand (no init pass).
// =====================================================================
__global__ __launch_bounds__(256) void lstm1_kernel(
    const float* __restrict__ x,       // (B, 1, T) == (B, T)
    const float* __restrict__ Wih_f, const float* __restrict__ Whh_f, const float* __restrict__ b_f,
    const float* __restrict__ Wih_b, const float* __restrict__ Whh_b, const float* __restrict__ b_b,
    float* __restrict__ out1)          // (B, T, 100): [fwd 0:50, bwd 50:100]
{
    const int H = 50, N = 200, KP = 64, NPG = 212, NT = 13, T = 128, NTHR = 256;

    __shared__ __bf16 Wb[208 * KP];    // [gate n][k] row-major, zero padded
    __shared__ __bf16 hb[16 * KP];     // h in bf16, zero padded
    __shared__ float  cb[16 * KP];     // c state
    __shared__ float  gb[16 * NPG];    // gate accumulators (f32), stride 212 (bank-safe)
    __shared__ float  xs[16 * T];      // x tile
    __shared__ float  wih_s[200];
    __shared__ float  bia_s[200];

    const int dir = blockIdx.x & 1;
    const int b0  = (blockIdx.x >> 1) * 16;
    const float* Wih = dir ? Wih_b : Wih_f;
    const float* Whh = dir ? Whh_b : Whh_f;
    const float* bia = dir ? b_b   : b_f;
    const int fofs = dir ? H : 0;
    const int tid = threadIdx.x;

    for (int idx = tid; idx < 208 * KP; idx += NTHR) {
        int n = idx / KP, k = idx % KP;
        Wb[idx] = (n < N && k < H) ? (__bf16)Whh[n * H + k] : (__bf16)0.0f;
    }
    for (int idx = tid; idx < 16 * KP; idx += NTHR) { hb[idx] = (__bf16)0.0f; cb[idx] = 0.0f; }
    for (int idx = tid; idx < 16 * T; idx += NTHR)  xs[idx] = x[(long)b0 * T + idx];
    for (int idx = tid; idx < N; idx += NTHR) { wih_s[idx] = Wih[idx]; bia_s[idx] = bia[idx]; }
    __syncthreads();

    const int lane = tid & 31, wave = tid >> 5;
    const int col = lane & 15, g = lane >> 4;

    for (int s = 0; s < T; ++s) {
        const int t = dir ? (T - 1 - s) : s;

        // gates = (x*Wih + b) + h @ Whh^T  via WMMA bf16 (C built in-register)
        float xv[8];
#pragma unroll
        for (int r = 0; r < 8; ++r) xv[r] = xs[(r + 8 * g) * T + t];

        for (int tile = wave; tile < NT; tile += 8) {
            const int n0 = tile * 16;
            const int n  = n0 + col;
            const float wn = (n < N) ? wih_s[n] : 0.0f;
            const float bn = (n < N) ? bia_s[n] : 0.0f;
            v8f acc;
#pragma unroll
            for (int r = 0; r < 8; ++r) acc[r] = xv[r] * wn + bn;
#pragma unroll
            for (int kb = 0; kb < 2; ++kb) {
                AF a;
                a.h[0] = *(const v8bf*)(hb + col * KP + kb * 32 + 16 * g);
                a.h[1] = *(const v8bf*)(hb + col * KP + kb * 32 + 16 * g + 8);
                v16bf bw = *(const v16bf*)(Wb + n * KP + kb * 32 + 16 * g);
                acc = __builtin_amdgcn_wmma_f32_16x16x32_bf16(
                    false, a.v, false, bw, (short)0, acc, false, false);
            }
#pragma unroll
            for (int r = 0; r < 8; ++r) gb[(r + 8 * g) * NPG + n0 + col] = acc[r];
        }
        __syncthreads();

        // elementwise LSTM cell
        for (int idx = tid; idx < 16 * H; idx += NTHR) {
            int m = idx / H, j = idx % H;
            float gi = gb[m * NPG + j];
            float gf = gb[m * NPG + H + j];
            float gg = gb[m * NPG + 2 * H + j];
            float go = gb[m * NPG + 3 * H + j];
            float c = sigf(gf) * cb[m * KP + j] + sigf(gi) * tanh_f(gg);
            float h = sigf(go) * tanh_f(c);
            cb[m * KP + j] = c;
            hb[m * KP + j] = (__bf16)h;
            out1[((long)(b0 + m) * T + t) * 100 + fofs + j] = h;
        }
        __syncthreads();
    }
}

// =====================================================================
// Input projection GEMM: out[M,600] = A[M,100] @ W[600,100]^T + bias
// One block = 16-row M tile, 8 waves cover 38 N tiles, 4 bf16 k-steps.
// kb 0..2 fully in-bounds -> aligned float4 loads; kb 3 bounds-checked.
// =====================================================================
__global__ __launch_bounds__(256) void xg_gemm_kernel(
    const float* __restrict__ A, long lda, int Mrows,
    const float* __restrict__ W, const float* __restrict__ bias,
    float* __restrict__ out)
{
    const int KIN = 100, NOUT = 600, NT = 38;
    const int m0 = blockIdx.x * 16;
    const int lane = threadIdx.x & 31, wave = threadIdx.x >> 5;
    const int col = lane & 15, g = lane >> 4;
    const int row = m0 + col;
    const float* Ar = A + (long)row * lda;
    (void)Mrows;

    for (int tile = wave; tile < NT; tile += 8) {
        const int n0 = tile * 16;
        const int n = n0 + col;
        v8f acc;
#pragma unroll
        for (int r = 0; r < 8; ++r) acc[r] = 0.0f;

#pragma unroll
        for (int kb = 0; kb < 3; ++kb) {                    // K in-bounds region
            const int kbase = kb * 32;
            AF a, bf;
            float av[16];
            *(float4*)&av[0]  = *(const float4*)(Ar + kbase + 16 * g);
            *(float4*)&av[4]  = *(const float4*)(Ar + kbase + 16 * g + 4);
            *(float4*)&av[8]  = *(const float4*)(Ar + kbase + 16 * g + 8);
            *(float4*)&av[12] = *(const float4*)(Ar + kbase + 16 * g + 12);
            float bv[16];
            if (n < NOUT) {
                const float* Wr = W + n * KIN + kbase + 16 * g;
                *(float4*)&bv[0]  = *(const float4*)(Wr);
                *(float4*)&bv[4]  = *(const float4*)(Wr + 4);
                *(float4*)&bv[8]  = *(const float4*)(Wr + 8);
                *(float4*)&bv[12] = *(const float4*)(Wr + 12);
            } else {
#pragma unroll
                for (int e = 0; e < 16; ++e) bv[e] = 0.0f;
            }
#pragma unroll
            for (int e = 0; e < 16; ++e) { a.e[e] = (__bf16)av[e]; bf.e[e] = (__bf16)bv[e]; }
            acc = __builtin_amdgcn_wmma_f32_16x16x32_bf16(
                false, a.v, false, bf.v, (short)0, acc, false, false);
        }
        {                                                   // kb == 3, K tail (96..99)
            const int kbase = 96;
            AF a, bf;
#pragma unroll
            for (int e = 0; e < 16; ++e) {
                int ka = kbase + 16 * g + e;
                a.e[e] = (ka < KIN) ? (__bf16)Ar[ka] : (__bf16)0.0f;
                int kn = kbase + 16 * g + e;
                bf.e[e] = (n < NOUT && kn < KIN) ? (__bf16)W[n * KIN + kn] : (__bf16)0.0f;
            }
            acc = __builtin_amdgcn_wmma_f32_16x16x32_bf16(
                false, a.v, false, bf.v, (short)0, acc, false, false);
        }
        if (n < NOUT) {
#pragma unroll
            for (int r = 0; r < 8; ++r)
                out[(long)(m0 + r + 8 * g) * NOUT + n] = acc[r] + bias[n];
        }
    }
}

// =====================================================================
// Layer-2 FORWARD recurrence. H2=150, Kpad=160 (5 k-steps), N=600 gates
// (38 tiles), 16 waves (512 thr) -> at most 3 tiles x 5 k-steps = 15
// serial WMMAs per timestep per wave. Whh bf16 [608][160] in dynamic
// LDS (~190KB). Per-timestep input gates double-buffered via CDNA5
// async Global->LDS copies (ASYNCcnt), hidden behind the WMMA +
// activation phases. Final h -> netf.  Total LDS = 288,256 B.
// =====================================================================
__global__ __launch_bounds__(512) void lstm2_kernel(
    const float* __restrict__ xg,      // (B, T, 600) precomputed forward gates
    const float* __restrict__ Whh,     // (600, 150)
    float* __restrict__ netf)          // (B, 300): writes [0:150)
{
    const int H = 150, N = 600, KP = 160, NPG = 612, NT = 38, T = 128, NTHR = 512;

    extern __shared__ char smem[];
    __bf16* Wb  = (__bf16*)smem;                       // 608*160 bf16   (194,560 B)
    __bf16* hb  = Wb + 608 * KP;                       // 16*160 bf16    (  5,120 B)
    float*  cb  = (float*)(hb + 16 * KP);              // 16*160 f32     ( 10,240 B)
    float*  gb0 = cb + 16 * KP;                        // 16*612 f32     ( 39,168 B)
    float*  gb1 = gb0 + 16 * NPG;                      // 16*612 f32     ( 39,168 B)

    const int b0 = blockIdx.x * 16;
    const int tid = threadIdx.x;

    if (tid == 0) { __builtin_prefetch(Whh, 0, 0); }   // global_prefetch_b8

    for (int idx = tid; idx < 608 * KP; idx += NTHR) {
        int n = idx / KP, k = idx % KP;
        Wb[idx] = (n < N && k < H) ? (__bf16)Whh[n * H + k] : (__bf16)0.0f;
    }
    for (int idx = tid; idx < 16 * KP; idx += NTHR) { hb[idx] = (__bf16)0.0f; cb[idx] = 0.0f; }

    // prologue: async-copy gates for t=0 into gb0 (150 x b128 per batch row)
    for (int idx = tid; idx < 2400; idx += NTHR) {
        int m = idx / 150, q = idx - m * 150;
        unsigned lds = (unsigned)(uintptr_t)(gb0 + m * NPG + q * 4);
        unsigned long long ga =
            (unsigned long long)(uintptr_t)(xg + ((long)(b0 + m) * T + 0) * (long)N + q * 4);
        asm volatile("global_load_async_to_lds_b128 %0, %1, off" :: "v"(lds), "v"(ga) : "memory");
    }
    asm volatile("s_wait_asynccnt 0" ::: "memory");
    __syncthreads();

    const int lane = tid & 31, wave = tid >> 5;
    const int col = lane & 15, g = lane >> 4;

    for (int t = 0; t < T; ++t) {
        float* gcur  = (t & 1) ? gb1 : gb0;
        float* gnext = (t & 1) ? gb0 : gb1;

        // kick off async copy of next timestep's gates (hidden behind compute)
        if (t + 1 < T) {
            for (int idx = tid; idx < 2400; idx += NTHR) {
                int m = idx / 150, q = idx - m * 150;
                unsigned lds = (unsigned)(uintptr_t)(gnext + m * NPG + q * 4);
                unsigned long long ga =
                    (unsigned long long)(uintptr_t)(xg + ((long)(b0 + m) * T + (t + 1)) * (long)N + q * 4);
                asm volatile("global_load_async_to_lds_b128 %0, %1, off" :: "v"(lds), "v"(ga) : "memory");
            }
        }

        // gates += h @ Whh^T via WMMA bf16
        for (int tile = wave; tile < NT; tile += 16) {
            const int n0 = tile * 16;
            v8f acc;
#pragma unroll
            for (int r = 0; r < 8; ++r) acc[r] = gcur[(r + 8 * g) * NPG + n0 + col];
#pragma unroll
            for (int kb = 0; kb < 5; ++kb) {
                AF a;
                a.h[0] = *(const v8bf*)(hb + col * KP + kb * 32 + 16 * g);
                a.h[1] = *(const v8bf*)(hb + col * KP + kb * 32 + 16 * g + 8);
                v16bf bw = *(const v16bf*)(Wb + (n0 + col) * KP + kb * 32 + 16 * g);
                acc = __builtin_amdgcn_wmma_f32_16x16x32_bf16(
                    false, a.v, false, bw, (short)0, acc, false, false);
            }
#pragma unroll
            for (int r = 0; r < 8; ++r) gcur[(r + 8 * g) * NPG + n0 + col] = acc[r];
        }
        __syncthreads();

        // elementwise LSTM cell
        for (int idx = tid; idx < 16 * H; idx += NTHR) {
            int m = idx / H, j = idx % H;
            float gi = gcur[m * NPG + j];
            float gf = gcur[m * NPG + H + j];
            float gg = gcur[m * NPG + 2 * H + j];
            float go = gcur[m * NPG + 3 * H + j];
            float c = sigf(gf) * cb[m * KP + j] + sigf(gi) * tanh_f(gg);
            float h = sigf(go) * tanh_f(c);
            cb[m * KP + j] = c;
            hb[m * KP + j] = (__bf16)h;
            if (t == T - 1) netf[(long)(b0 + m) * 300 + j] = h;
        }
        asm volatile("s_wait_asynccnt 0" ::: "memory");    // next-step gates landed
        __syncthreads();
    }
}

// =====================================================================
// Tail: layer-2 backward single step (reverse scan output at t=T-1 is
// its first step, h0=c0=0) + FC heads. One block per batch sample.
// =====================================================================
__global__ __launch_bounds__(64) void tail_kernel(
    const float* __restrict__ xgb,     // (B, 600) backward gates at t=T-1
    const float* __restrict__ netf,    // (B, 300) forward half filled
    const float* __restrict__ rrs,     // (B, 4)
    const float* __restrict__ headW, const float* __restrict__ headb,
    const float* __restrict__ pef1W, const float* __restrict__ pef1b,
    const float* __restrict__ pef2W, const float* __restrict__ pef2b,
    const float* __restrict__ fcW,   const float* __restrict__ fcb,
    float* __restrict__ out)
{
    const int B20 = 1024 * 20, B2 = 1024 * 2, B30 = 1024 * 30;
    const int OFF_PL = B20, OFF_FEAT = B20 + B2, OFF_LOG = B20 + B2 + B30;

    __shared__ float net[300];
    __shared__ float wf[20];
    __shared__ float pef[10];

    const int b = blockIdx.x;
    const int tid = threadIdx.x;

    for (int j = tid; j < 150; j += 64) {
        net[j] = netf[(long)b * 300 + j];
        float gi = xgb[(long)b * 600 + j];
        float gg = xgb[(long)b * 600 + 300 + j];
        float go = xgb[(long)b * 600 + 450 + j];
        float c = sigf(gi) * tanh_f(gg);           // f-gate * c0 term is zero
        net[150 + j] = sigf(go) * tanh_f(c);
    }
    __syncthreads();

    if (tid < 20) {
        float s = headb[tid];
        for (int j = 0; j < 300; ++j) s += net[j] * headW[tid * 300 + j];
        wf[tid] = leaky(s);
    } else if (tid >= 32 && tid < 42) {
        int u = tid - 32;
        float s = pef1b[u];
        for (int k = 0; k < 4; ++k) s += rrs[(long)b * 4 + k] * pef1W[u * 4 + k];
        pef[u] = leaky(s);
    }
    __syncthreads();

    if (tid < 20) out[(long)b * 20 + tid] = wf[tid];
    if (tid < 2) {
        float s = pef2b[tid];
        for (int k = 0; k < 10; ++k) s += pef[k] * pef2W[tid * 10 + k];
        out[OFF_PL + (long)b * 2 + tid] = s;
    }
    if (tid < 30) out[OFF_FEAT + (long)b * 30 + tid] = (tid < 20) ? wf[tid] : pef[tid - 20];
    if (tid < 4) {
        float s = fcb[tid];
        for (int i = 0; i < 20; ++i) s += wf[i] * fcW[tid * 30 + i];
        for (int i = 0; i < 10; ++i) s += pef[i] * fcW[tid * 30 + 20 + i];
        out[OFF_LOG + (long)b * 4 + tid] = s;
    }
}

// =====================================================================
extern "C" void kernel_launch(void* const* d_in, const int* in_sizes, int n_in,
                              void* d_out, int out_size, void* d_ws, size_t ws_size,
                              hipStream_t stream) {
    (void)in_sizes; (void)n_in; (void)out_size; (void)ws_size;
    const float* x       = (const float*)d_in[0];
    const float* rrs     = (const float*)d_in[1];
    const float* l1Wih_f = (const float*)d_in[2];
    const float* l1Whh_f = (const float*)d_in[3];
    const float* l1b_f   = (const float*)d_in[4];
    const float* l1Wih_b = (const float*)d_in[5];
    const float* l1Whh_b = (const float*)d_in[6];
    const float* l1b_b   = (const float*)d_in[7];
    const float* l2Wih_f = (const float*)d_in[8];
    const float* l2Whh_f = (const float*)d_in[9];
    const float* l2b_f   = (const float*)d_in[10];
    const float* l2Wih_b = (const float*)d_in[11];
    // d_in[12] = l2_Whh_b (unused: backward dir only runs 1 step from h0=0)
    const float* l2b_b   = (const float*)d_in[13];
    const float* headW   = (const float*)d_in[14];
    const float* headb   = (const float*)d_in[15];
    const float* pef1W   = (const float*)d_in[16];
    const float* pef1b   = (const float*)d_in[17];
    const float* pef2W   = (const float*)d_in[18];
    const float* pef2b   = (const float*)d_in[19];
    const float* fcW     = (const float*)d_in[20];
    const float* fcb     = (const float*)d_in[21];
    float* out = (float*)d_out;

    const int B = 1024, T = 128;
    // workspace carve (bytes, all 256-aligned)
    char* ws = (char*)d_ws;
    float* out1 = (float*)(ws);                                  // B*T*100 f32  (52.4 MB)
    float* xg2f = (float*)(ws + (size_t)52428800);               // B*T*600 f32  (314.6 MB)
    float* xg2b = (float*)(ws + (size_t)367001600);              // B*600  f32   (2.4 MB)
    float* netf = (float*)(ws + (size_t)369459200);              // B*300  f32

    // 1) Layer-1 BiLSTM (both dirs), WMMA recurrence
    lstm1_kernel<<<dim3(B / 16 * 2), dim3(256), 0, stream>>>(
        x, l1Wih_f, l1Whh_f, l1b_f, l1Wih_b, l1Whh_b, l1b_b, out1);

    // 2) Layer-2 input projections (WMMA GEMM)
    //    forward: all timesteps;  backward: only t=T-1 row per batch
    xg_gemm_kernel<<<dim3(B * T / 16), dim3(256), 0, stream>>>(
        out1, (long)100, B * T, l2Wih_f, l2b_f, xg2f);
    xg_gemm_kernel<<<dim3(B / 16), dim3(256), 0, stream>>>(
        out1 + (long)(T - 1) * 100, (long)T * 100, B, l2Wih_b, l2b_b, xg2b);

    // 3) Layer-2 forward recurrence (Whh bf16 resident in LDS, async-
    //    double-buffered gate tiles)
    const int LSTM2_LDS = (608 * 160 + 16 * 160) * 2 + (16 * 160 + 2 * 16 * 612) * 4; // 288256 B
    (void)hipFuncSetAttribute(reinterpret_cast<const void*>(lstm2_kernel),
                              hipFuncAttributeMaxDynamicSharedMemorySize, LSTM2_LDS);
    lstm2_kernel<<<dim3(B / 16), dim3(512), LSTM2_LDS, stream>>>(xg2f, l2Whh_f, netf);

    // 4) Layer-2 backward single step + FC heads + outputs
    tail_kernel<<<dim3(B), dim3(64), 0, stream>>>(
        xg2b, netf, rrs, headW, headb, pef1W, pef1b, pef2W, pef2b, fcW, fcb, out);
}
